// Model_3D_28269474742575
// MI455X (gfx1250) — compile-verified, matrix operand order checked by reading, not run on hardware
//
#include <hip/hip_runtime.h>
#include <hip/hip_bf16.h>

typedef unsigned short u16;
typedef __attribute__((ext_vector_type(16))) __bf16 bf16x16;
typedef __attribute__((ext_vector_type(8)))  float  v8f;

#define DEVFN __device__ __forceinline__

DEVFN u16 f2bf(float f) {
  union { float f; unsigned u; } x; x.f = f;
  unsigned r = x.u + 0x7FFFu + ((x.u >> 16) & 1u);
  return (u16)(r >> 16);
}
DEVFN float bf2f(u16 u) {
  union { unsigned u; float f; } x; x.u = ((unsigned)u) << 16;
  return x.f;
}
DEVFN float gelu_f(float x) { return 0.5f * x * (1.0f + erff(x * 0.70710678118654752f)); }

union FragBF {
  bf16x16 v;
  u16 u[16];
  uint4 q[2];
};

DEVFN v8f vzero() { v8f z = {0.f,0.f,0.f,0.f,0.f,0.f,0.f,0.f}; return z; }

DEVFN v8f wmma_bf16(const FragBF& a, const FragBF& b, v8f c) {
  // D = A(16x32 bf16) x B(32x16 bf16) + C(f32)
  return __builtin_amdgcn_wmma_f32_16x16x32_bf16(false, a.v, false, b.v, (short)0, c, false, false);
}

// ---------------------------------------------------------------------------
// Generic bf16 WMMA GEMM: Out[M,Nout] = act(A[M,K] * W[Nout,K]^T + bias) (+R)
// Requirements: M % 64 == 0, Nout % 32 == 0, K % 32 == 0 (true for all uses).
// Block: 256 threads = 8 waves (2 M x 4 N); macro-tile 64(M) x 128(N).
// Wave tile 32x32: 4 WMMAs / K-step. K-loop is software-pipelined with
// double-buffered fragments so next-step loads fly under current WMMAs.
// Batched over blockIdx.z with (outer, inner) offsets, z = zb*zHdiv + zh.
// ---------------------------------------------------------------------------
struct GemmP {
  const u16* A;  long lda, aOut, aIn;
  const u16* W;  long ldw, wOut, wIn;
  const float* bias;
  const float* R; long ldr, rOut, rIn;
  float* oF; long ldoF, oFOut, oFIn;
  u16*   oB; long ldoB, oBOut, oBIn;
  int M, K, Nout, act, zHdiv;
};

__global__ void __launch_bounds__(256) gemm_bf16_wmma(GemmP p) {
  int z  = blockIdx.z;
  int zb = z / p.zHdiv, zh = z % p.zHdiv;
  const u16* A = p.A + (long)zb * p.aOut + (long)zh * p.aIn;
  const u16* W = p.W + (long)zb * p.wOut + (long)zh * p.wIn;

  int wave = threadIdx.x >> 5, lane = threadIdx.x & 31;
  int half = lane >> 4, r = lane & 15;
  int m0 = blockIdx.x * 64  + (wave & 1) * 32;
  int n0 = blockIdx.y * 128 + (wave >> 1) * 32;
  if (m0 >= p.M || n0 >= p.Nout) return;   // wave-uniform

  v8f acc00 = vzero(), acc01 = vzero(), acc10 = vzero(), acc11 = vzero();
  const u16* pa0 = A + (long)(m0 + r) * p.lda + half * 8;
  const u16* pa1 = pa0 + 16 * p.lda;
  const u16* pw0 = W + (long)(n0 + r) * p.ldw + half * 8;
  const u16* pw1 = pw0 + 16 * p.ldw;

  FragBF a0A, a1A, b0A, b1A, a0B, a1B, b0B, b1B;
  auto LD = [&](FragBF& a0, FragBF& a1, FragBF& b0, FragBF& b1) {
    a0.q[0] = *(const uint4*)(pa0); a0.q[1] = *(const uint4*)(pa0 + 16);
    a1.q[0] = *(const uint4*)(pa1); a1.q[1] = *(const uint4*)(pa1 + 16);
    b0.q[0] = *(const uint4*)(pw0); b0.q[1] = *(const uint4*)(pw0 + 16);
    b1.q[0] = *(const uint4*)(pw1); b1.q[1] = *(const uint4*)(pw1 + 16);
    pa0 += 32; pa1 += 32; pw0 += 32; pw1 += 32;
    __builtin_prefetch(pa0 + 64, 0, 1);   // global_prefetch_b8, 2 K-steps past the next load
  };
  auto CP = [&](FragBF& a0, FragBF& a1, FragBF& b0, FragBF& b1) {
    acc00 = wmma_bf16(a0, b0, acc00);
    acc01 = wmma_bf16(a0, b1, acc01);
    acc10 = wmma_bf16(a1, b0, acc10);
    acc11 = wmma_bf16(a1, b1, acc11);
  };

  int s = p.K >> 5;                 // number of 32-deep K steps (>=1)
  LD(a0A, a1A, b0A, b1A);
  int i = 0;
  for (; i + 2 < s; i += 2) {       // ping-pong: load next while computing cur
    LD(a0B, a1B, b0B, b1B); CP(a0A, a1A, b0A, b1A);
    LD(a0A, a1A, b0A, b1A); CP(a0B, a1B, b0B, b1B);
  }
  if (i + 1 < s) {
    LD(a0B, a1B, b0B, b1B); CP(a0A, a1A, b0A, b1A);
    CP(a0B, a1B, b0B, b1B);
  } else {
    CP(a0A, a1A, b0A, b1A);
  }

  const float* Rp = p.R  ? p.R  + (long)zb * p.rOut  + (long)zh * p.rIn  : nullptr;
  float* oF = p.oF ? p.oF + (long)zb * p.oFOut + (long)zh * p.oFIn : nullptr;
  u16*   oB = p.oB ? p.oB + (long)zb * p.oBOut + (long)zh * p.oBIn : nullptr;

  v8f accs[2][2] = {{acc00, acc01}, {acc10, acc11}};
  #pragma unroll
  for (int mi = 0; mi < 2; ++mi) {
    #pragma unroll
    for (int si = 0; si < 2; ++si) {
      v8f acc = accs[mi][si];
      int n = n0 + si * 16 + r;
      float bs = p.bias ? p.bias[n] : 0.f;
      #pragma unroll
      for (int v = 0; v < 8; ++v) {
        int m = m0 + mi * 16 + v + 8 * half;
        float val = acc[v] + bs;
        if (p.act == 1) val = gelu_f(val);
        if (Rp) val += Rp[(long)m * p.ldr + n];
        if (oF) oF[(long)m * p.ldoF + n] = val;
        if (oB) oB[(long)m * p.ldoB + n] = f2bf(val);
      }
    }
  }
}

// ---------------------------------------------------------------------------
// Implicit-GEMM 3x3x3 conv over 32^3 grid, C=256 -> 256, SAME padding.
// Input/output [B, N=32768, C=256] bf16 (channel-last). Weight [27][O][I] bf16.
// Block: 256 threads = 8 waves (2 n x 4 o); macro-tile 64(n) x 128(o).
// Wave tile 32x32 with per-row-group border masks on the A loads only.
// 8 K-steps per tap, software-pipelined (double-buffered fragments).
// ---------------------------------------------------------------------------
__global__ void __launch_bounds__(256) conv3d_bf16_wmma(
    const u16* __restrict__ Y, const u16* __restrict__ Wt,
    const float* __restrict__ bias, u16* __restrict__ out) {
  int b = blockIdx.z;
  int wave = threadIdx.x >> 5, lane = threadIdx.x & 31;
  int half = lane >> 4, r = lane & 15;
  int m0 = blockIdx.x * 64  + (wave & 1) * 32;
  int o0 = blockIdx.y * 128 + (wave >> 1) * 32;

  int n_0 = m0 + r,      n_1 = m0 + 16 + r;
  int cz0 = n_0 & 31, by0 = (n_0 >> 5) & 31, ax0 = n_0 >> 10;
  int cz1 = n_1 & 31, by1 = (n_1 >> 5) & 31, ax1 = n_1 >> 10;

  v8f acc00 = vzero(), acc01 = vzero(), acc10 = vzero(), acc11 = vzero();
  for (int tap = 0; tap < 27; ++tap) {
    int da = tap / 9 - 1, db = (tap / 3) % 3 - 1, dc = tap % 3 - 1;
    int aa0 = ax0 + da, bb0 = by0 + db, cc0 = cz0 + dc;
    int aa1 = ax1 + da, bb1 = by1 + db, cc1 = cz1 + dc;
    bool valid0 = ((unsigned)aa0 < 32u) && ((unsigned)bb0 < 32u) && ((unsigned)cc0 < 32u);
    bool valid1 = ((unsigned)aa1 < 32u) && ((unsigned)bb1 < 32u) && ((unsigned)cc1 < 32u);
    const u16* src0 = Y + ((long)b * 32768 + (((aa0 * 32) + bb0) * 32 + cc0)) * 256 + half * 8;
    const u16* src1 = Y + ((long)b * 32768 + (((aa1 * 32) + bb1) * 32 + cc1)) * 256 + half * 8;
    const u16* wt0 = Wt + (long)tap * 65536 + (long)(o0 + r) * 256 + half * 8;
    const u16* wt1 = wt0 + 16 * 256;

    FragBF a0A, a1A, w0A, w1A, a0B, a1B, w0B, w1B;
    auto LDc = [&](FragBF& a0, FragBF& a1, FragBF& w0, FragBF& w1) {
      if (valid0) { a0.q[0] = *(const uint4*)(src0); a0.q[1] = *(const uint4*)(src0 + 16); }
      else        { a0.q[0] = make_uint4(0,0,0,0);   a0.q[1] = make_uint4(0,0,0,0); }
      if (valid1) { a1.q[0] = *(const uint4*)(src1); a1.q[1] = *(const uint4*)(src1 + 16); }
      else        { a1.q[0] = make_uint4(0,0,0,0);   a1.q[1] = make_uint4(0,0,0,0); }
      w0.q[0] = *(const uint4*)(wt0); w0.q[1] = *(const uint4*)(wt0 + 16);
      w1.q[0] = *(const uint4*)(wt1); w1.q[1] = *(const uint4*)(wt1 + 16);
      src0 += 32; src1 += 32; wt0 += 32; wt1 += 32;
    };
    auto CPc = [&](FragBF& a0, FragBF& a1, FragBF& w0, FragBF& w1) {
      acc00 = wmma_bf16(a0, w0, acc00);
      acc01 = wmma_bf16(a0, w1, acc01);
      acc10 = wmma_bf16(a1, w0, acc10);
      acc11 = wmma_bf16(a1, w1, acc11);
    };

    // 8 K-steps, pipelined: LD; 3x(LD,CP,LD,CP); LD,CP; CP
    LDc(a0A, a1A, w0A, w1A);
    #pragma unroll
    for (int it = 0; it < 3; ++it) {
      LDc(a0B, a1B, w0B, w1B); CPc(a0A, a1A, w0A, w1A);
      LDc(a0A, a1A, w0A, w1A); CPc(a0B, a1B, w0B, w1B);
    }
    LDc(a0B, a1B, w0B, w1B); CPc(a0A, a1A, w0A, w1A);
    CPc(a0B, a1B, w0B, w1B);
  }

  v8f accs[2][2] = {{acc00, acc01}, {acc10, acc11}};
  #pragma unroll
  for (int mi = 0; mi < 2; ++mi) {
    #pragma unroll
    for (int si = 0; si < 2; ++si) {
      v8f acc = accs[mi][si];
      int o = o0 + si * 16 + r;
      float bs = bias[o];
      #pragma unroll
      for (int v = 0; v < 8; ++v) {
        int m = m0 + mi * 16 + v + 8 * half;
        out[((long)b * 32768 + m) * 256 + o] = f2bf(acc[v] + bs);
      }
    }
  }
}

// ---------------------------------------------------------------------------
// st[b,h,g,c] = (1/(snorm+1e-5)) * sum_n swT[b,h,g,n] * fx_mid[b,n,h*32+c]
// Per (b,h): M=32 (g), Nout=32 (c), K=32768 (n). 2 waves, wave tile 16x32.
// ---------------------------------------------------------------------------
__global__ void __launch_bounds__(64) st_bf16_wmma(
    const u16* __restrict__ swT, const u16* __restrict__ fxm,
    const float* __restrict__ snorm, u16* __restrict__ st) {
  int bh = blockIdx.x; int b = bh >> 3, hh = bh & 7;
  int wave = threadIdx.x >> 5, lane = threadIdx.x & 31;
  int half = lane >> 4, r = lane & 15;
  int g0 = wave * 16;

  v8f acc0 = vzero(), acc1 = vzero();
  const u16* pa = swT + ((long)bh * 32 + g0 + r) * 32768 + half * 8;
  const u16* fbase = fxm + (long)b * 32768 * 256 + hh * 32;
  for (int k0 = 0; k0 < 32768; k0 += 32) {
    FragBF a, f0, f1;
    a.q[0] = *(const uint4*)(pa); a.q[1] = *(const uint4*)(pa + 16);
    pa += 32;
    #pragma unroll
    for (int e = 0; e < 16; ++e) {
      int nn = k0 + ((e < 8) ? (half * 8 + e) : (16 + half * 8 + (e - 8)));
      const u16* fp = fbase + (long)nn * 256;
      f0.u[e] = fp[r];
      f1.u[e] = fp[16 + r];
    }
    acc0 = wmma_bf16(a, f0, acc0);
    acc1 = wmma_bf16(a, f1, acc1);
  }
  for (int s2 = 0; s2 < 2; ++s2) {
    v8f acc = s2 ? acc1 : acc0;
    #pragma unroll
    for (int v = 0; v < 8; ++v) {
      int g = g0 + v + 8 * half;
      float d = snorm[bh * 32 + g] + 1e-5f;
      st[((long)bh * 32 + g) * 32 + s2 * 16 + r] = f2bf(acc[v] / d);
    }
  }
}

// 32x32x32 GEMM helper: acc[mi][ni] += A-tile x W-tile^T (both row-major [32,32])
DEVFN void mm32(const u16* A, int lda, const u16* W, int ldw, int half, int r, v8f acc[2][2]) {
  for (int mi = 0; mi < 2; ++mi) {
    FragBF a;
    const u16* pa = A + (mi * 16 + r) * lda + half * 8;
    a.q[0] = *(const uint4*)pa; a.q[1] = *(const uint4*)(pa + 16);
    for (int ni = 0; ni < 2; ++ni) {
      FragBF w;
      const u16* pw = W + (ni * 16 + r) * ldw + half * 8;
      w.q[0] = *(const uint4*)pw; w.q[1] = *(const uint4*)(pw + 16);
      acc[mi][ni] = wmma_bf16(a, w, acc[mi][ni]);
    }
  }
}
DEVFN void zacc(v8f a[2][2]) { for (int i = 0; i < 2; ++i) for (int j = 0; j < 2; ++j) a[i][j] = vzero(); }

// ---------------------------------------------------------------------------
// Per-(b,h) tiny attention: q/k/v = st@W^T; attn = softmax(q k^T / sqrt(32));
// oT[c,g] = (attn @ v)^T. One wave per block, LDS round-trips between WMMAs.
// ---------------------------------------------------------------------------
__global__ void __launch_bounds__(32) attn_core(
    const u16* __restrict__ st, const u16* __restrict__ wq, const u16* __restrict__ wk,
    const u16* __restrict__ wv, u16* __restrict__ oT) {
  __shared__ __align__(16) u16 qs[1024];
  __shared__ __align__(16) u16 ks[1024];
  __shared__ __align__(16) u16 vts[1024];
  __shared__ __align__(16) u16 ats[1024];
  __shared__ float ss[1024];

  int bh = blockIdx.x;
  int lane = threadIdx.x & 31, half = lane >> 4, r = lane & 15;
  const u16* S = st + (long)bh * 1024;
  v8f acc[2][2];

  zacc(acc); mm32(S, 32, wq, 32, half, r, acc);
  for (int mi = 0; mi < 2; ++mi) for (int ni = 0; ni < 2; ++ni)
    for (int v = 0; v < 8; ++v) {
      int m = mi * 16 + v + 8 * half, nn = ni * 16 + r;
      qs[m * 32 + nn] = f2bf(acc[mi][ni][v]);
    }
  zacc(acc); mm32(S, 32, wk, 32, half, r, acc);
  for (int mi = 0; mi < 2; ++mi) for (int ni = 0; ni < 2; ++ni)
    for (int v = 0; v < 8; ++v) {
      int m = mi * 16 + v + 8 * half, nn = ni * 16 + r;
      ks[m * 32 + nn] = f2bf(acc[mi][ni][v]);
    }
  zacc(acc); mm32(S, 32, wv, 32, half, r, acc);
  for (int mi = 0; mi < 2; ++mi) for (int ni = 0; ni < 2; ++ni)
    for (int v = 0; v < 8; ++v) {
      int m = mi * 16 + v + 8 * half, nn = ni * 16 + r;
      vts[nn * 32 + m] = f2bf(acc[mi][ni][v]);          // v transposed
    }
  __syncthreads();

  zacc(acc); mm32(qs, 32, ks, 32, half, r, acc);        // s = q k^T
  for (int mi = 0; mi < 2; ++mi) for (int ni = 0; ni < 2; ++ni)
    for (int v = 0; v < 8; ++v) {
      int m = mi * 16 + v + 8 * half, nn = ni * 16 + r;
      ss[m * 32 + nn] = acc[mi][ni][v] * 0.17677669529663687f;  // 32^-0.5
    }
  __syncthreads();

  {
    int row = lane;
    float mx = -1e30f;
    #pragma unroll
    for (int j = 0; j < 32; ++j) mx = fmaxf(mx, ss[row * 32 + j]);
    float e[32]; float sum = 0.f;
    #pragma unroll
    for (int j = 0; j < 32; ++j) { e[j] = expf(ss[row * 32 + j] - mx); sum += e[j]; }
    float inv = 1.f / sum;
    #pragma unroll
    for (int j = 0; j < 32; ++j) ats[row * 32 + j] = f2bf(e[j] * inv);
  }
  __syncthreads();

  zacc(acc); mm32(ats, 32, vts, 32, half, r, acc);      // o = attn @ v
  for (int mi = 0; mi < 2; ++mi) for (int ni = 0; ni < 2; ++ni)
    for (int v = 0; v < 8; ++v) {
      int g = mi * 16 + v + 8 * half, c = ni * 16 + r;
      oT[(long)bh * 1024 + c * 32 + g] = f2bf(acc[mi][ni][v]);   // stored transposed
    }
}

// ---------------------------------------------------------------------------
// Elementwise / helper kernels
// ---------------------------------------------------------------------------
__global__ void cvt_bf16_kernel(const float* __restrict__ s, u16* __restrict__ d, long n) {
  long i = (long)blockIdx.x * 256 + threadIdx.x;
  if (i < n) d[i] = f2bf(s[i]);
}
__global__ void conv_reorder_kernel(const float* __restrict__ s, u16* __restrict__ d) {
  int i = blockIdx.x * 256 + threadIdx.x;     // over 27*256*256
  int t = i / 65536, oi = i % 65536;
  int o = oi >> 8, ii = oi & 255;
  d[i] = f2bf(s[(o * 256 + ii) * 27 + t]);    // [O,I,27] -> [27][O][I]
}
__global__ void zero_f32_kernel(float* p, int n) {
  int i = blockIdx.x * 256 + threadIdx.x;
  if (i < n) p[i] = 0.f;
}
__global__ void pre1_kernel(const float* __restrict__ x, const float* __restrict__ fx,
                            const float* __restrict__ w1, const float* __restrict__ b1,
                            u16* __restrict__ h1) {
  long i = (long)blockIdx.x * 256 + threadIdx.x;   // over BN*512
  long m = i >> 9; int j = (int)(i & 511);
  float a = w1[j*4+0]*x[m*3+0] + w1[j*4+1]*x[m*3+1] + w1[j*4+2]*x[m*3+2] + w1[j*4+3]*fx[m] + b1[j];
  h1[i] = f2bf(gelu_f(a));
}
// One wave per row; lane holds 8 channels; wave32 shuffle reduction.
__global__ void __launch_bounds__(256) ln_kernel(const float* __restrict__ h,
    const float* __restrict__ g, const float* __restrict__ bta, u16* __restrict__ y, int rows) {
  int wave = threadIdx.x >> 5, lane = threadIdx.x & 31;
  int row = blockIdx.x * 8 + wave;
  if (row >= rows) return;
  const float* p = h + (long)row * 256;
  float v[8]; float s = 0.f;
  #pragma unroll
  for (int j = 0; j < 8; ++j) { v[j] = p[lane + 32 * j]; s += v[j]; }
  for (int m = 16; m >= 1; m >>= 1) s += __shfl_xor(s, m, 32);
  float mean = s * (1.f / 256.f);
  float vs = 0.f;
  #pragma unroll
  for (int j = 0; j < 8; ++j) { float d = v[j] - mean; vs += d * d; }
  for (int m = 16; m >= 1; m >>= 1) vs += __shfl_xor(vs, m, 32);
  float rstd = rsqrtf(vs * (1.f / 256.f) + 1e-5f);
  #pragma unroll
  for (int j = 0; j < 8; ++j) {
    int c = lane + 32 * j;
    y[(long)row * 256 + c] = f2bf((v[j] - mean) * rstd * g[c] + bta[c]);
  }
}
// Softmax over G=32 per (b,h,n): writes sw, swT, accumulates snorm via LDS+global atomics.
__global__ void __launch_bounds__(256) softmax_kernel(const float* __restrict__ logits,
    const float* __restrict__ temp, u16* __restrict__ sw, u16* __restrict__ swT,
    float* __restrict__ snorm) {
  __shared__ float red[32];
  int tid = threadIdx.x;
  long rho = (long)blockIdx.x * 256 + tid;     // (b*8+h)*N + n
  int bh = (int)(rho >> 15);
  int n = (int)(rho & 32767);
  int h = bh & 7;
  float t = fminf(fmaxf(temp[h], 0.1f), 5.f);
  float l[32]; float mx = -1e30f;
  #pragma unroll
  for (int j = 0; j < 32; ++j) { l[j] = logits[rho * 32 + j] / t; mx = fmaxf(mx, l[j]); }
  float s = 0.f;
  #pragma unroll
  for (int j = 0; j < 32; ++j) { l[j] = expf(l[j] - mx); s += l[j]; }
  float inv = 1.f / s;
  if (tid < 32) red[tid] = 0.f;
  __syncthreads();
  #pragma unroll
  for (int j = 0; j < 32; ++j) {
    float w = l[j] * inv;
    sw[rho * 32 + j] = f2bf(w);
    swT[((long)bh * 32 + j) * 32768 + n] = f2bf(w);
    atomicAdd(&red[j], w);
  }
  __syncthreads();
  if (tid < 32) atomicAdd(&snorm[bh * 32 + tid], red[tid]);
}
__global__ void head_kernel(const u16* __restrict__ y3, const float* __restrict__ wo2,
                            const float* __restrict__ bo2, float* __restrict__ out) {
  long i = (long)blockIdx.x * 256 + threadIdx.x;   // BN
  float a = bo2[0];
  const u16* p = y3 + i * 256;
  #pragma unroll 8
  for (int c = 0; c < 256; ++c) a += bf2f(p[c]) * wo2[c];
  out[i] = a;
}

// ---------------------------------------------------------------------------
extern "C" void kernel_launch(void* const* d_in, const int* in_sizes, int n_in,
                              void* d_out, int out_size, void* d_ws, size_t ws_size,
                              hipStream_t stream) {
  (void)in_sizes; (void)n_in; (void)out_size; (void)ws_size;
  const long N = 32768, BN = 65536;
  auto F = [&](int i) { return (const float*)d_in[i]; };

  char* ws = (char*)d_ws;
  const size_t Mi = 1ull << 20;
  float* h      = (float*)(ws + 0);          // [BN,256] f32 residual  (64Mi)
  u16*   y      = (u16*)(ws + 64 * Mi);      // LN output bf16         (32Mi)
  u16*   h1     = (u16*)(ws + 96 * Mi);      // pre-MLP mid bf16       (shared 64Mi)
  float* logits = (float*)(ws + 96 * Mi);    // [B,H,N,G] f32          (shared 64Mi)
  u16*   fxm    = (u16*)(ws + 160 * Mi);     // fx_mid / mlp-mid bf16  (32Mi)
  u16*   xm     = (u16*)(ws + 192 * Mi);     // x_mid bf16             (32Mi)
  u16*   sw     = (u16*)(ws + 224 * Mi);     // [B,H,N,G] bf16         (32Mi)
  u16*   swT    = (u16*)(ws + 256 * Mi);     // [B,H,G,N] bf16         (32Mi)
  u16*   oxb    = (u16*)(ws + 288 * Mi);     // ox bf16                (32Mi)
  char*  wp     = ws + 320 * Mi;
  u16* pw2b  = (u16*)wp; wp += 1 * Mi;
  u16* wfxT  = (u16*)wp; wp += 4 * Mi;
  u16* wxT   = (u16*)wp; wp += 4 * Mi;
  u16* wslb  = (u16*)wp; wp += 64 * 1024;
  u16* wqb   = (u16*)wp; wp += 64 * 1024;
  u16* wkb   = (u16*)wp; wp += 64 * 1024;
  u16* wvb   = (u16*)wp; wp += 64 * 1024;
  u16* woutb = (u16*)wp; wp += 1 * Mi;
  u16* wm1b  = (u16*)wp; wp += 1 * Mi;
  u16* wm2b  = (u16*)wp; wp += 1 * Mi;
  float* snorm = (float*)wp; wp += 64 * 1024;
  u16* stb   = (u16*)wp; wp += 64 * 1024;
  u16* oTb   = (u16*)wp; wp += 64 * 1024;

  auto cvt = [&](const float* s, u16* d, long n) {
    cvt_bf16_kernel<<<dim3((unsigned)((n + 255) / 256)), 256, 0, stream>>>(s, d, n);
  };
  auto gemm = [&](GemmP p, int zc) {
    dim3 grid((p.M + 63) / 64, (p.Nout + 127) / 128, zc);
    gemm_bf16_wmma<<<grid, 256, 0, stream>>>(p);
  };

  // ---- pre stage: h = gelu(cat(x,fx)@W1^T+b1)@W2^T+b2 ----
  cvt(F(4), pw2b, 256 * 512);
  pre1_kernel<<<dim3((unsigned)(BN * 512 / 256)), 256, 0, stream>>>(F(0), F(1), F(2), F(3), h1);
  {
    GemmP p{}; p.A = h1; p.lda = 512; p.W = pw2b; p.ldw = 512; p.bias = F(5);
    p.oF = h; p.ldoF = 256; p.M = (int)BN; p.K = 512; p.Nout = 256; p.zHdiv = 1;
    gemm(p, 1);
  }

  for (int blk = 0; blk < 5; ++blk) {
    int bse = 6 + blk * 20;
    // weight conversion (per block; scratch reused)
    conv_reorder_kernel<<<dim3(27 * 65536 / 256), 256, 0, stream>>>(F(bse + 3), wfxT);
    conv_reorder_kernel<<<dim3(27 * 65536 / 256), 256, 0, stream>>>(F(bse + 5), wxT);
    cvt(F(bse + 7), wslb, 1024);
    cvt(F(bse + 9), wqb, 1024);  cvt(F(bse + 10), wkb, 1024);  cvt(F(bse + 11), wvb, 1024);
    cvt(F(bse + 12), woutb, 65536);
    cvt(F(bse + 16), wm1b, 65536);
    cvt(F(bse + 18), wm2b, 65536);

    // LN1 -> y (bf16)
    ln_kernel<<<dim3((unsigned)(BN / 8)), 256, 0, stream>>>(h, F(bse + 0), F(bse + 1), y, (int)BN);
    // convs (implicit GEMM, WMMA)
    conv3d_bf16_wmma<<<dim3(512, 2, 2), 256, 0, stream>>>(y, wfxT, F(bse + 4), fxm);
    conv3d_bf16_wmma<<<dim3(512, 2, 2), 256, 0, stream>>>(y, wxT,  F(bse + 6), xm);
    // slice logits per (b,h): [N,32] = x_mid_h[N,32] @ w_slice^T + b_slice
    {
      GemmP p{}; p.A = xm; p.lda = 256; p.aOut = N * 256; p.aIn = 32;
      p.W = wslb; p.ldw = 32; p.bias = F(bse + 8);
      p.oF = logits; p.ldoF = 32; p.oFOut = 8L * N * 32; p.oFIn = N * 32;
      p.M = (int)N; p.K = 32; p.Nout = 32; p.zHdiv = 8;
      gemm(p, 16);
    }
    zero_f32_kernel<<<2, 256, 0, stream>>>(snorm, 512);
    softmax_kernel<<<dim3((unsigned)(16L * N / 256)), 256, 0, stream>>>(logits, F(bse + 2), sw, swT, snorm);
    st_bf16_wmma<<<16, 64, 0, stream>>>(swT, fxm, snorm, stb);
    attn_core<<<16, 32, 0, stream>>>(stb, wqb, wkb, wvb, oTb);
    // ox per (b,h): [N,32] = sw[N,32] @ oT^T  (oT stored [c,g])
    {
      GemmP p{}; p.A = sw; p.lda = 32; p.aOut = 8L * N * 32; p.aIn = N * 32;
      p.W = oTb; p.ldw = 32; p.wOut = 8L * 1024; p.wIn = 1024;
      p.oB = oxb; p.ldoB = 256; p.oBOut = N * 256; p.oBIn = 32;
      p.M = (int)N; p.K = 32; p.Nout = 32; p.zHdiv = 8;
      gemm(p, 16);
    }
    // w_out projection + residual -> h
    {
      GemmP p{}; p.A = oxb; p.lda = 256; p.W = woutb; p.ldw = 256; p.bias = F(bse + 13);
      p.R = h; p.ldr = 256; p.oF = h; p.ldoF = 256;
      p.M = (int)BN; p.K = 256; p.Nout = 256; p.zHdiv = 1;
      gemm(p, 1);
    }
    // MLP: LN2 -> gelu GEMM -> GEMM + residual
    ln_kernel<<<dim3((unsigned)(BN / 8)), 256, 0, stream>>>(h, F(bse + 14), F(bse + 15), y, (int)BN);
    {
      GemmP p{}; p.A = y; p.lda = 256; p.W = wm1b; p.ldw = 256; p.bias = F(bse + 17);
      p.act = 1; p.oB = fxm; p.ldoB = 256;
      p.M = (int)BN; p.K = 256; p.Nout = 256; p.zHdiv = 1;
      gemm(p, 1);
    }
    {
      GemmP p{}; p.A = fxm; p.lda = 256; p.W = wm2b; p.ldw = 256; p.bias = F(bse + 19);
      p.R = h; p.ldr = 256; p.oF = h; p.ldoF = 256;
      p.M = (int)BN; p.K = 256; p.Nout = 256; p.zHdiv = 1;
      gemm(p, 1);
    }
  }

  // final head: LN3 then [BN,256]x[256,1]
  ln_kernel<<<dim3((unsigned)(BN / 8)), 256, 0, stream>>>(h, F(106), F(107), y, (int)BN);
  head_kernel<<<dim3((unsigned)(BN / 256)), 256, 0, stream>>>(y, F(108), F(109), (float*)d_out);
}
